// FastGroupSelfAttention_58497454572127
// MI455X (gfx1250) — matrix operand order repaired and checked by reading, not compiled, not run on hardware
//
#include <hip/hip_runtime.h>

// ---- CDNA5 wave32 WMMA types -------------------------------------------------
typedef float        v8f   __attribute__((ext_vector_type(8)));
typedef __bf16       v16bf __attribute__((ext_vector_type(16)));
typedef unsigned int u32x4 __attribute__((ext_vector_type(4)));
typedef int          i32x4 __attribute__((ext_vector_type(4)));
typedef int          i32x8 __attribute__((ext_vector_type(8)));

#if __has_builtin(__builtin_amdgcn_tensor_load_to_lds)
#define HAS_TDM 1
#else
#define HAS_TDM 0
#endif

#if HAS_TDM
#define IF_TDM(x) x
#else
#define IF_TDM(x)
#endif

// A-matrix (16x32 bf16) fragment: lanes 0-15 hold row m=lane, halfwords [0..15]=K0..15;
// lanes 16-31 hold K16..31  ->  koff = (lane>>4)*8, start = 2*koff.
static __device__ __forceinline__ v16bf load_af(const __bf16* p, int koff) {
    union { v16bf v; int4 q[2]; } u;
    u.q[0] = *(const int4*)(p + 2 * koff);
    u.q[1] = *(const int4*)(p + 2 * koff + 8);
    return u.v;
}
// B-matrix (32x16 bf16) fragment: 16 contiguous halfwords per lane.
static __device__ __forceinline__ v16bf load_bf32(const __bf16* p) {
    union { v16bf v; int4 q[2]; } u;
    u.q[0] = *(const int4*)(p);
    u.q[1] = *(const int4*)(p + 8);
    return u.v;
}

#if HAS_TDM
// Issue a TDM 2D tile load: `rows` rows of `rowlen` bf16 elements, global row
// stride `rstride` (elements), into LDS byte offset `lds_off`.
// D# per CDNA5 ISA ch.8: group0 = {count/flags, lds_addr, global_addr, type=2},
// group1 = {data_size, tensor_dim0/1, tile_dim0/1/2, tensor_dim0_stride}.
// clang-23 form: (u32x4, i32x8, i32x4, i32x4, i32x8, i32 cpol).
static __device__ __forceinline__ void tdm_load_tile(unsigned lds_off,
                                                     const void* gptr,
                                                     unsigned rows,
                                                     unsigned rowlen,
                                                     unsigned long long rstride) {
    unsigned long long ga = (unsigned long long)gptr;
    u32x4 g0;
    g0[0] = 1u;                                           // count=1, user D#
    g0[1] = lds_off;                                      // lds_addr (bytes)
    g0[2] = (unsigned)(ga & 0xffffffffu);                 // global_addr[31:0]
    g0[3] = (unsigned)((ga >> 32) & 0x01ffffffu) | (2u << 30);  // [56:32] | type=2
    i32x8 g1;
    g1[0] = (int)(1u << 16);                              // data_size = 2 bytes
    g1[1] = (int)((rowlen & 0xffffu) << 16);              // tensor_dim0[15:0]
    g1[2] = (int)((rowlen >> 16) | ((rows & 0xffffu) << 16)); // tdim0_hi|tdim1_lo
    g1[3] = (int)((rows >> 16) | ((rowlen & 0xffffu) << 16)); // tdim1_hi|tile_dim0
    g1[4] = (int)(rows & 0xffffu);                        // tile_dim1 (tile_dim2=0)
    g1[5] = (int)(rstride & 0xffffffffu);                 // tensor_dim0_stride lo
    g1[6] = (int)((rstride >> 32) & 0xffffu);             // stride hi | tdim1_stride=0
    g1[7] = 0;
    i32x4 z4 = {0, 0, 0, 0};
    i32x8 z8 = {0, 0, 0, 0, 0, 0, 0, 0};
    __builtin_amdgcn_tensor_load_to_lds(g0, g1, z4, z4, z8, 0);
}
#endif

// ---- Kernel 0a: pack Wq/Wk/Wv -> bf16 WMMA-B layout, and Wo ------------------
extern "C" __global__ __launch_bounds__(256)
void k_pack_w(const float* __restrict__ Wq, const float* __restrict__ Wk,
              const float* __restrict__ Wv, const float* __restrict__ Wo,
              __bf16* __restrict__ wqkv, __bf16* __restrict__ wop, int D) {
    const int KT = D >> 5;
    size_t idx = (size_t)blockIdx.x * blockDim.x + threadIdx.x;
    size_t nqkv = (size_t)3 * D * D;
    if (idx < nqkv) {
        int j    = (int)(idx & 15);
        int lane = (int)((idx >> 4) & 31);
        size_t rest = idx >> 9;              // nt*KT + kt
        int kt = (int)(rest % KT);
        int nt = (int)(rest / KT);
        int k  = kt * 32 + ((lane >> 4) << 4) + j;
        int n  = nt * 16 + (lane & 15);
        const float* W = (n < D) ? Wq : ((n < 2 * D) ? Wk : Wv);
        int nc = n - (n / D) * D;
        wqkv[idx] = (__bf16)W[(size_t)k * D + nc];
    }
    if (idx < (size_t)D * D) {
        int j    = (int)(idx & 15);
        int lane = (int)((idx >> 4) & 31);
        size_t rest = idx >> 9;
        int kt = (int)(rest % KT);
        int nt = (int)(rest / KT);
        int k  = kt * 32 + ((lane >> 4) << 4) + j;
        int n  = nt * 16 + (lane & 15);
        wop[idx] = (__bf16)Wo[(size_t)k * D + n];
    }
}

// ---- Kernel 0b: rowmap[b] = real_flat_idx[unscatter_idx[b]] ------------------
extern "C" __global__ void k_rowmap(const int* __restrict__ rflat,
                                    const int* __restrict__ unsc,
                                    int* __restrict__ rowmap, int B) {
    int b = blockIdx.x * blockDim.x + threadIdx.x;
    if (b < B) rowmap[b] = rflat[unsc[b]];
}

// ---- Kernel 1: RMSNorm + gather to packed (T, G*M, D) bf16 -------------------
extern "C" __global__ __launch_bounds__(256)
void k_rmsnorm_pack(const float* __restrict__ hidden, const float* __restrict__ lnw,
                    const int* __restrict__ gip, __bf16* __restrict__ packed,
                    int T, int D, int GM, int M) {
    const int t = blockIdx.x, g = blockIdx.y;
    const int wid = threadIdx.x >> 5, lane = threadIdx.x & 31;
    for (int m = wid; m < M; m += 8) {
        const int b = gip[g * M + m];
        const float* src = hidden + ((size_t)b * T + t) * D;
        float ss = 0.f;
        for (int c = lane; c < D; c += 32) { float v = src[c]; ss += v * v; }
#pragma unroll
        for (int o = 16; o >= 1; o >>= 1) ss += __shfl_xor(ss, o, 32);
        const float r = rsqrtf(ss / (float)D + 1e-6f);
        __bf16* dst = packed + ((size_t)t * GM + (size_t)g * M + m) * D;
        for (int c = lane; c < D; c += 32)
            dst[c] = (__bf16)(src[c] * r * lnw[c]);
    }
}

// ============================================================================
// GEMM core: block = 8 waves covering 32 rows x 256 cols.  The B panel of each
// k-step (16 n-tiles x 512 bf16, already in WMMA-B fragment order) is staged
// into LDS by the Tensor Data Mover (double buffered), while waves run WMMAs
// on the previous buffer.  Sync: s_wait_tensorcnt (all waves; only wave0 has
// outstanding TDM ops) + s_barrier.  The input-operand asm keeps sB's contents
// "written by an unknown writer" (the TDM) without hiding the pointer's LDS
// address space, so the fragment reads stay ds_load_b128.
// ============================================================================
#define GEMM_PIPE_BODY(A_ROW_PTR, WP, KT, ACC, LANE, CSUB, KOFF)                  \
    __shared__ __bf16 sB[2][16 * 512]; /* 2 x 16KB double buffer */               \
    {                                                                             \
        const int wid_ = __builtin_amdgcn_readfirstlane(threadIdx.x >> 5);        \
        const unsigned long long rstride = (unsigned long long)(KT) * 512ull;     \
        const __bf16* gB0 = (WP);                                                 \
        (void)wid_; (void)rstride;                                                \
        IF_TDM(if (wid_ == 0) tdm_load_tile(0u, gB0, 16u, 512u, rstride);)        \
        _Pragma("unroll 1")                                                       \
        for (int kt = 0; kt < (KT); ++kt) {                                       \
            if (HAS_TDM) {                                                        \
                IF_TDM(                                                           \
                if (kt + 1 < (KT)) {                                              \
                    if (wid_ == 0)                                                \
                        tdm_load_tile(((kt + 1) & 1) ? 16384u : 0u,               \
                                      gB0 + (size_t)(kt + 1) * 512, 16u, 512u,    \
                                      rstride);                                   \
                    __builtin_amdgcn_s_wait_tensorcnt(1);                         \
                } else {                                                          \
                    __builtin_amdgcn_s_wait_tensorcnt(0);                         \
                } )                                                               \
            } else {                                                              \
                /* fallback: cooperative synchronous stage of this k-step */      \
                const __bf16* src = gB0 + (size_t)kt * 512;                       \
                for (int q = threadIdx.x; q < 16 * 512 / 8; q += 256) {           \
                    int ntl = q >> 6, e8 = (q & 63) << 3;                         \
                    *(int4*)&sB[kt & 1][ntl * 512 + e8] =                         \
                        *(const int4*)(src + (size_t)ntl * (KT) * 512 + e8);      \
                }                                                                 \
            }                                                                     \
            __syncthreads();                                                      \
            v16bf a = load_af((A_ROW_PTR) + kt * 32, (KOFF));                     \
            const __bf16* bb = &sB[kt & 1][((CSUB) * 4) * 512 + (LANE) * 16];     \
            /* TDM wrote sB behind the compiler's back: unknown-writer fence */   \
            asm volatile("" : : "v"(bb) : "memory");                              \
            v16bf b0 = load_bf32(bb);                                             \
            v16bf b1 = load_bf32(bb + 512);                                       \
            v16bf b2 = load_bf32(bb + 1024);                                      \
            v16bf b3 = load_bf32(bb + 1536);                                      \
            (ACC)[0] = __builtin_amdgcn_wmma_f32_16x16x32_bf16(                   \
                false, a, false, b0, (short)0, (ACC)[0], false, false);           \
            (ACC)[1] = __builtin_amdgcn_wmma_f32_16x16x32_bf16(                   \
                false, a, false, b1, (short)0, (ACC)[1], false, false);           \
            (ACC)[2] = __builtin_amdgcn_wmma_f32_16x16x32_bf16(                   \
                false, a, false, b2, (short)0, (ACC)[2], false, false);           \
            (ACC)[3] = __builtin_amdgcn_wmma_f32_16x16x32_bf16(                   \
                false, a, false, b3, (short)0, (ACC)[3], false, false);           \
            __syncthreads();                                                      \
        }                                                                         \
    }

// ---- Kernel 2: fused QKV GEMM  (N x D) * (D x 3D) -> bf16 --------------------
extern "C" __global__ __launch_bounds__(256)
void k_qkv_gemm(const __bf16* __restrict__ A, const __bf16* __restrict__ Wp,
                __bf16* __restrict__ QKV, long N, int D) {
    const int C3 = 3 * D;
    const int KT = D >> 5;
    const int wid = threadIdx.x >> 5, lane = threadIdx.x & 31;
    const int rsub = wid >> 2, csub = wid & 3;
    long rowA = (long)blockIdx.x * 32 + rsub * 16 + (lane & 15);
    if (rowA >= N) rowA = N - 1;
    const __bf16* arow = A + rowA * D;
    const int ntbase = blockIdx.y * 16 + csub * 4;
    const int koff = (lane >> 4) << 3;
    // block's B panel: 16 n-tiles starting at blockIdx.y*16, one k-tile each step
    const __bf16* wpanel = Wp + ((size_t)blockIdx.y * 16) * (size_t)KT * 512;
    v8f acc[4] = {};
    GEMM_PIPE_BODY(arow, wpanel, KT, acc, lane, csub, koff)
    const int n = lane & 15, mb = (lane >> 4) << 3;
#pragma unroll
    for (int j = 0; j < 4; ++j) {
        int col = (ntbase + j) * 16 + n;
        if (col >= C3) continue;
#pragma unroll
        for (int r = 0; r < 8; ++r) {
            long m = (long)blockIdx.x * 32 + rsub * 16 + mb + r;
            if (m < N) QKV[m * C3 + col] = (__bf16)acc[j][r];
        }
    }
}

// ---- Kernel 3: per-(t,g) attention, wave-per-head ----------------------------
extern "C" __global__ __launch_bounds__(256)
void k_attn(const __bf16* __restrict__ QKV, const float* __restrict__ pmask,
            __bf16* __restrict__ CTX, int T, int D, int GM, int M) {
    __shared__ __bf16 s_vt[8][64][32];   // V^T per wave's head, zero padded to K=32
    __shared__ float  s_sc[8][32][33];   // raw scores / exp scratch
    __shared__ __bf16 s_at[8][32][32];   // attn probs as WMMA A operand
    const int t = blockIdx.x, g = blockIdx.y;
    const int wid = threadIdx.x >> 5, lane = threadIdx.x & 31;
    const int H = D >> 6;
    const int C3 = 3 * D;
    int Mt = (M + 15) >> 4; if (Mt > 2) Mt = 2;   // supports M <= 32
    const int Mpad = Mt << 4;
    const long rowbase = (long)t * GM + (long)g * M;
    const long NROW = (long)T * GM;
    const int koff = (lane >> 4) << 3;
    const int kb   = (lane >> 4) << 4;

    for (int h = wid; h < H; h += 8) {
        // stage V^T into LDS (rows >= M zeroed)
        for (int idx = lane; idx < 64 * 32; idx += 32) {
            int e = idx >> 5, n = idx & 31;
            __bf16 v = (__bf16)0.f;
            if (n < M) v = QKV[(rowbase + n) * C3 + 2 * D + h * 64 + e];
            s_vt[wid][e][n] = v;
        }
        // scores = Q * K^T   (HEAD_D = 64 -> two K=32 WMMAs)
        for (int i = 0; i < Mt; ++i)
            for (int j = 0; j < Mt; ++j) {
                long rq = rowbase + i * 16 + (lane & 15); if (rq >= NROW) rq = NROW - 1;
                long rk = rowbase + j * 16 + (lane & 15); if (rk >= NROW) rk = NROW - 1;
                v16bf a0 = load_af(QKV + rq * C3 + h * 64, koff);
                v16bf b0 = load_bf32(QKV + rk * C3 + D + h * 64 + kb);
                v16bf a1 = load_af(QKV + rq * C3 + h * 64 + 32, koff);
                v16bf b1 = load_bf32(QKV + rk * C3 + D + h * 64 + 32 + kb);
                v8f c = {};
                c = __builtin_amdgcn_wmma_f32_16x16x32_bf16(
                    false, a0, false, b0, (short)0, c, false, false);
                c = __builtin_amdgcn_wmma_f32_16x16x32_bf16(
                    false, a1, false, b1, (short)0, c, false, false);
                const int n = lane & 15, mb = (lane >> 4) << 3;
#pragma unroll
                for (int r = 0; r < 8; ++r)
                    s_sc[wid][i * 16 + mb + r][j * 16 + n] = c[r];
            }
        // masked softmax: lane == row
        {
            const int m = lane;
            if (m < M) {
                float mx = -3.4028235e38f;
                for (int n = 0; n < Mpad; ++n) {
                    float s = (n < M)
                        ? s_sc[wid][m][n] + pmask[((size_t)g * M + m) * M + n]
                        : -3.4028235e38f;
                    s_sc[wid][m][n] = s;
                    mx = fmaxf(mx, s);
                }
                float sum = 0.f;
                for (int n = 0; n < Mpad; ++n) {
                    float e = __expf(s_sc[wid][m][n] - mx);
                    s_sc[wid][m][n] = e;
                    sum += e;
                }
                const float inv = 1.f / sum;
                for (int n = 0; n < 32; ++n)
                    s_at[wid][m][n] = (n < Mpad) ? (__bf16)(s_sc[wid][m][n] * inv)
                                                 : (__bf16)0.f;
            } else if (m < 32) {
                for (int n = 0; n < 32; ++n) s_at[wid][m][n] = (__bf16)0.f;
            }
        }
        // ctx = attn * V   (single K=32 WMMA per 16-col tile of HEAD_D)
        for (int i = 0; i < Mt; ++i) {
            v16bf a = load_af(&s_at[wid][i * 16 + (lane & 15)][0], koff);
#pragma unroll
            for (int et = 0; et < 4; ++et) {
                v16bf b = load_bf32(&s_vt[wid][et * 16 + (lane & 15)][kb]);
                v8f c = {};
                c = __builtin_amdgcn_wmma_f32_16x16x32_bf16(
                    false, a, false, b, (short)0, c, false, false);
                const int e = lane & 15, mb = (lane >> 4) << 3;
#pragma unroll
                for (int r = 0; r < 8; ++r) {
                    int m = i * 16 + mb + r;
                    if (m < M)
                        CTX[(rowbase + m) * D + h * 64 + et * 16 + e] = (__bf16)c[r];
                }
            }
        }
    }
}

// ---- Kernel 4: out = hidden + gather(ctx) * Wo  (fp32 output, (B,T,D)) -------
extern "C" __global__ __launch_bounds__(256)
void k_out_gemm(const __bf16* __restrict__ CTX, const __bf16* __restrict__ WoP,
                const float* __restrict__ hidden, const int* __restrict__ rowmap,
                float* __restrict__ out, int T, int D, int GM, int B) {
    const int KT = D >> 5;
    const int wid = threadIdx.x >> 5, lane = threadIdx.x & 31;
    const int rsub = wid >> 2, csub = wid & 3;
    const int t = blockIdx.z;
    int b0 = blockIdx.x * 32 + rsub * 16 + (lane & 15);
    if (b0 >= B) b0 = B - 1;
    const long row = (long)t * GM + rowmap[b0];
    const __bf16* arow = CTX + row * D;
    const int ntbase = blockIdx.y * 16 + csub * 4;
    const int koff = (lane >> 4) << 3;
    const __bf16* wpanel = WoP + ((size_t)blockIdx.y * 16) * (size_t)KT * 512;
    v8f acc[4] = {};
    GEMM_PIPE_BODY(arow, wpanel, KT, acc, lane, csub, koff)
    const int n = lane & 15, mb = (lane >> 4) << 3;
#pragma unroll
    for (int j = 0; j < 4; ++j) {
        int col = (ntbase + j) * 16 + n;
        if (col >= D) continue;
#pragma unroll
        for (int r = 0; r < 8; ++r) {
            int b = blockIdx.x * 32 + rsub * 16 + mb + r;
            if (b < B) {
                size_t o = ((size_t)b * T + t) * D + col;
                out[o] = hidden[o] + acc[j][r];
            }
        }
    }
}

// ---- host launcher -----------------------------------------------------------
extern "C" void kernel_launch(void* const* d_in, const int* in_sizes, int n_in,
                              void* d_out, int out_size, void* d_ws, size_t ws_size,
                              hipStream_t stream) {
    (void)n_in; (void)out_size; (void)ws_size;
    const float* hidden = (const float*)d_in[0];
    const float* lnw    = (const float*)d_in[2];
    const float* Wq     = (const float*)d_in[3];
    const float* Wk     = (const float*)d_in[4];
    const float* Wv     = (const float*)d_in[5];
    const float* Wo     = (const float*)d_in[6];
    const int*   gip    = (const int*)d_in[7];
    const float* pmask  = (const float*)d_in[8];
    const int*   rflat  = (const int*)d_in[9];
    const int*   unsc   = (const int*)d_in[10];
    float* out = (float*)d_out;

    const int D  = in_sizes[2];
    const int GM = in_sizes[7];
    const int M  = in_sizes[8] / in_sizes[7];
    const int G  = GM / M;
    const int B  = in_sizes[9];
    const int T  = (int)((size_t)in_sizes[0] / ((size_t)B * D));

    char* ws = (char*)d_ws;
    size_t off = 0;
    auto carve = [&](size_t bytes) -> char* {
        char* p = ws + off;
        off += (bytes + 255) & ~(size_t)255;
        return p;
    };
    __bf16* wqkv   = (__bf16*)carve((size_t)3 * D * D * sizeof(__bf16));
    __bf16* wop    = (__bf16*)carve((size_t)D * D * sizeof(__bf16));
    int*    rowmap = (int*)carve((size_t)B * sizeof(int));
    __bf16* packed = (__bf16*)carve((size_t)T * GM * D * sizeof(__bf16));
    __bf16* qkv    = (__bf16*)carve((size_t)T * GM * 3 * D * sizeof(__bf16));
    __bf16* ctx    = (__bf16*)carve((size_t)T * GM * D * sizeof(__bf16));

    {   // weight pack (covers both wqkv and wo)
        long nel = (long)3 * D * D;
        unsigned blk = (unsigned)((nel + 255) / 256);
        k_pack_w<<<dim3(blk), dim3(256), 0, stream>>>(Wq, Wk, Wv, Wo, wqkv, wop, D);
    }
    k_rowmap<<<dim3((B + 255) / 256), dim3(256), 0, stream>>>(rflat, unsc, rowmap, B);
    k_rmsnorm_pack<<<dim3(T, G), dim3(256), 0, stream>>>(hidden, lnw, gip, packed,
                                                         T, D, GM, M);
    {   // QKV projection
        long N = (long)T * GM;
        unsigned gx = (unsigned)((N + 31) / 32);
        unsigned gy = (unsigned)((3 * D + 255) / 256);
        k_qkv_gemm<<<dim3(gx, gy), dim3(256), 0, stream>>>(packed, wqkv, qkv, N, D);
    }
    k_attn<<<dim3(T, G), dim3(256), 0, stream>>>(qkv, pmask, ctx, T, D, GM, M);
    {   // output projection + scatter + residual
        unsigned gx = (unsigned)((B + 31) / 32);
        unsigned gy = (unsigned)((D + 255) / 256);
        k_out_gemm<<<dim3(gx, gy, T), dim3(256), 0, stream>>>(ctx, wop, hidden,
                                                              rowmap, out, T, D, GM, B);
    }
}
